// NoPEAttention_53764400611743
// MI455X (gfx1250) — matrix operand order repaired and checked by reading, not compiled, not run on hardware
//
#include <hip/hip_runtime.h>
#include <hip/hip_bf16.h>

// Problem constants (from reference)
#define S_LEN 2048
#define HID   4096
#define NH    32
#define NKV   8
#define HD    128

typedef __attribute__((ext_vector_type(16))) __bf16        v16bf;
typedef __attribute__((ext_vector_type(8)))  float          v8f;
typedef __attribute__((ext_vector_type(8)))  unsigned short u16x8;
typedef __attribute__((ext_vector_type(4)))  float          f32x4;
typedef __attribute__((ext_vector_type(4)))  int            i32x4;

#define AS1 __attribute__((address_space(1)))
#define AS3 __attribute__((address_space(3)))

#if __has_builtin(__builtin_amdgcn_global_load_async_to_lds_b128)
#define HAVE_ASYNC_LDS 1
#else
#define HAVE_ASYNC_LDS 0
#endif

union BF16Frag { v16bf v; unsigned short u[16]; u16x8 h[2]; };

__device__ __forceinline__ unsigned short f32_to_bf16(float f) {
  union { float f; unsigned u; } x; x.f = f;
  unsigned u = x.u + 0x7FFFu + ((x.u >> 16) & 1u);   // round-to-nearest-even
  return (unsigned short)(u >> 16);
}

__device__ __forceinline__ u16x8 ld8u(const unsigned short* p) {
  return *(const u16x8*)p;
}

__device__ __forceinline__ v8f wmma_bf16(v16bf a, v16bf b, v8f c) {
  // D = A(16x32 bf16) * B(32x16 bf16) + C(16x16 f32)
  return __builtin_amdgcn_wmma_f32_16x16x32_bf16(false, a, false, b, (short)0, c,
                                                 false, false);
}

__device__ __forceinline__ void async_copy_b128(const void* g, void* l) {
#if HAVE_ASYNC_LDS
  // signature (probe-derived): (int4 addrspace(1)*, int4 addrspace(3)*, imm offset, imm cpol)
  __builtin_amdgcn_global_load_async_to_lds_b128((AS1 i32x4*)g, (AS3 i32x4*)l, 0, 0);
#endif
}

__device__ __forceinline__ void wait_async_zero() {
#if HAVE_ASYNC_LDS
#if __has_builtin(__builtin_amdgcn_s_wait_asynccnt)
  __builtin_amdgcn_s_wait_asynccnt(0);
#else
  asm volatile("s_wait_asynccnt 0" ::: "memory");
#endif
#endif
}

// ---------------------------------------------------------------------------
// Elementwise f32 -> bf16 conversion (bandwidth-bound, runs once per tensor).
// ---------------------------------------------------------------------------
__global__ __launch_bounds__(256) void cvt_f32_bf16(const float* __restrict__ s,
                                                    unsigned short* __restrict__ d,
                                                    int n) {
  const int i = (blockIdx.x * 256 + threadIdx.x) * 8;
  if (i + 8 <= n) {
    f32x4 a = *(const f32x4*)(s + i);
    f32x4 b = *(const f32x4*)(s + i + 4);
    u16x8 o;
#pragma unroll
    for (int j = 0; j < 4; ++j) { o[j] = f32_to_bf16(a[j]); o[4 + j] = f32_to_bf16(b[j]); }
    *(u16x8*)(d + i) = o;
  }
}

// ---------------------------------------------------------------------------
// All-bf16 NT GEMM: Y[M,N] = A[M,K] * W[N,K]^T, K = 4096.
// Block = 256 thr = 8 waves; block tile 256(M) x 64(N); wave tile 32 x 64.
// W tile (64x32) staged in LDS per K-step (async global->LDS on CDNA5).
// ---------------------------------------------------------------------------
template <bool OUTF32>
__global__ __launch_bounds__(256) void gemm_bf16_nt(const unsigned short* __restrict__ A,
                                                    const unsigned short* __restrict__ W,
                                                    void* __restrict__ Yp,
                                                    int M, int N) {
  __shared__ unsigned short Ws[64 * 40];   // [n-local][k-local], padded rows

  const int K    = HID;
  const int tid  = threadIdx.x;
  const int widx = tid >> 5;
  const int lane = tid & 31;
  const int lo   = lane & 15;
  const int hi   = lane >> 4;

  const int nb  = N >> 6;
  const int bm  = blockIdx.x / nb;
  const int bn  = blockIdx.x % nb;
  const int n0  = bn * 64;
  const int m0  = bm * 256 + widx * 32;

  const unsigned short* Ar0 = A + (size_t)(m0 + lo) * K;
  const unsigned short* Ar1 = Ar0 + (size_t)16 * K;

  // staging coords: each thread moves 8 contiguous bf16 (one b128)
  const int sr = tid >> 2;          // 0..63  (n-local row)
  const int sc = (tid & 3) * 8;     // 0,8,16,24 (k-local col)
  const unsigned short* wsrc0 = W + (size_t)(n0 + sr) * K + sc;
  unsigned short* wdst = &Ws[sr * 40 + sc];

  v8f acc0[4] = {}, acc1[4] = {};

  for (int kk = 0; kk < K; kk += 32) {
    __syncthreads();
#if HAVE_ASYNC_LDS
    async_copy_b128(wsrc0 + kk, wdst);
    wait_async_zero();
#else
    *(u16x8*)wdst = ld8u(wsrc0 + kk);
#endif
    __syncthreads();

    // A fragments: rows m0+lo and m0+16+lo; K chunks [hi*8,+8) and [16+hi*8,+8)
    BF16Frag fa0, fa1;
    fa0.h[0] = ld8u(Ar0 + kk + hi * 8);
    fa0.h[1] = ld8u(Ar0 + kk + 16 + hi * 8);
    fa1.h[0] = ld8u(Ar1 + kk + hi * 8);
    fa1.h[1] = ld8u(Ar1 + kk + 16 + hi * 8);
    __builtin_prefetch(Ar0 + kk + 64, 0, 0);
    __builtin_prefetch(Ar1 + kk + 64, 0, 0);

#pragma unroll
    for (int t = 0; t < 4; ++t) {
      const unsigned short* wr = &Ws[(t * 16 + lo) * 40 + hi * 8];
      BF16Frag fb;
      fb.h[0] = ld8u(wr);
      fb.h[1] = ld8u(wr + 16);
      acc0[t] = wmma_bf16(fa0.v, fb.v, acc0[t]);
      acc1[t] = wmma_bf16(fa1.v, fb.v, acc1[t]);
    }
  }

  // C layout: VGPR r -> row +r+8*hi, lane.lo -> column
#pragma unroll
  for (int t = 0; t < 4; ++t) {
#pragma unroll
    for (int r = 0; r < 8; ++r) {
      const int row0 = m0 + r + 8 * hi;
      const int col  = n0 + t * 16 + lo;
      if constexpr (OUTF32) {
        ((float*)Yp)[(size_t)row0 * N + col]        = acc0[t][r];
        ((float*)Yp)[(size_t)(row0 + 16) * N + col] = acc1[t][r];
      } else {
        ((unsigned short*)Yp)[(size_t)row0 * N + col]        = f32_to_bf16(acc0[t][r]);
        ((unsigned short*)Yp)[(size_t)(row0 + 16) * N + col] = f32_to_bf16(acc1[t][r]);
      }
    }
  }
}

// ---------------------------------------------------------------------------
// Flash attention (causal, GQA 4:1), bf16 in/out.
// Grid: (S/128 query blocks, NH heads). Block: 256 threads = 8 waves,
// wave w owns query rows [qb*128 + w*16, +16).
// ---------------------------------------------------------------------------
__global__ __launch_bounds__(256) void attn_fwd(const unsigned short* __restrict__ Q,
                                                const unsigned short* __restrict__ Kg,
                                                const unsigned short* __restrict__ Vg,
                                                unsigned short* __restrict__ O) {
  __shared__ unsigned short Ks[32 * 136];   // K tile [key][d], padded
  __shared__ unsigned short Vt[128 * 40];   // V tile transposed [d][key], padded
  __shared__ unsigned short Ps[8 * 16 * 40];// per-wave P bounce buffer

  const int h    = blockIdx.y;
  const int qb   = blockIdx.x;
  const int kvh  = h >> 2;                  // repeat_interleave(groups=4)
  const int tid  = threadIdx.x;
  const int w    = tid >> 5;
  const int lane = tid & 31;
  const int lo   = lane & 15;
  const int hi   = lane >> 4;
  const int qrow0 = qb * 128 + w * 16;

  // Q fragments (A-matrix, 4 d-chunks of 32), resident for whole kernel
  BF16Frag qa[4];
  {
    const unsigned short* qp = Q + (size_t)(qrow0 + lo) * (NH * HD) + h * HD;
#pragma unroll
    for (int dc = 0; dc < 4; ++dc) {
      qa[dc].h[0] = ld8u(qp + dc * 32 + hi * 8);
      qa[dc].h[1] = ld8u(qp + dc * 32 + 16 + hi * 8);
    }
  }

  v8f o[8] = {};
  float mrun[8], lrun[8];
#pragma unroll
  for (int r = 0; r < 8; ++r) { mrun[r] = -3.0e38f; lrun[r] = 0.f; }

  const int   ktiles = qb * 4 + 4;          // keys up to qb*128+127 (causal)
  const float scale  = 0.08838834764831845f; // 1/sqrt(128)

  // staging coords
  const int kr_ = (tid * 16) >> 7, kc_ = (tid * 16) & 127;       // K tile
  const int vr_ = tid >> 3,        vc_ = (tid & 7) * 16;         // V tile

  for (int t = 0; t < ktiles; ++t) {
    const int k0 = t * 32;
    __syncthreads();
    { // stage K tile 32x128 -> LDS (async path on CDNA5)
      const unsigned short* src = Kg + (size_t)(k0 + kr_) * (NKV * HD) + kvh * HD + kc_;
      unsigned short* dst = &Ks[kr_ * 136 + kc_];
#if HAVE_ASYNC_LDS
      async_copy_b128(src, dst);
      async_copy_b128(src + 8, dst + 8);
#else
      *(u16x8*)dst       = ld8u(src);
      *(u16x8*)(dst + 8) = ld8u(src + 8);
#endif
    }
    { // stage V tile 32x128 transposed (d-major) so PV B-frags load contiguous
      const unsigned short* src = Vg + (size_t)(k0 + vr_) * (NKV * HD) + kvh * HD + vc_;
      u16x8 x0 = ld8u(src), x1 = ld8u(src + 8);
#pragma unroll
      for (int i = 0; i < 8; ++i) {
        Vt[(vc_ + i) * 40 + vr_]     = x0[i];
        Vt[(vc_ + 8 + i) * 40 + vr_] = x1[i];
      }
    }
    wait_async_zero();
    __syncthreads();

    // scores: S(16x32) = Q(16x128) * K_tile^T, 2 N-halves x 4 d-chunks
    v8f sc[2] = {};
#pragma unroll
    for (int j = 0; j < 2; ++j) {
      const unsigned short* kr = &Ks[(j * 16 + lo) * 136];
#pragma unroll
      for (int dc = 0; dc < 4; ++dc) {
        BF16Frag kb;
        kb.h[0] = ld8u(kr + dc * 32 + hi * 8);
        kb.h[1] = ld8u(kr + dc * 32 + 16 + hi * 8);
        sc[j] = wmma_bf16(qa[dc].v, kb.v, sc[j]);
      }
    }

    // online softmax (rows live across the 16 lanes of a half-wave)
    float pr0[8], pr1[8];
#pragma unroll
    for (int r = 0; r < 8; ++r) {
      const int rowg = qrow0 + r + 8 * hi;
      const int key0 = k0 + lo, key1 = k0 + 16 + lo;
      float a0 = (key0 <= rowg) ? sc[0][r] * scale : -1.0e30f;
      float a1 = (key1 <= rowg) ? sc[1][r] * scale : -1.0e30f;
      float tm = fmaxf(a0, a1);
#pragma unroll
      for (int m = 1; m < 16; m <<= 1) tm = fmaxf(tm, __shfl_xor(tm, m, 32));
      const float mnew  = fmaxf(mrun[r], tm);
      const float alpha = __expf(mrun[r] - mnew);
      const float p0 = __expf(a0 - mnew);
      const float p1 = __expf(a1 - mnew);
      float rs = p0 + p1;
#pragma unroll
      for (int m = 1; m < 16; m <<= 1) rs += __shfl_xor(rs, m, 32);
      lrun[r] = lrun[r] * alpha + rs;
      mrun[r] = mnew;
      pr0[r] = p0; pr1[r] = p1;
#pragma unroll
      for (int dt = 0; dt < 8; ++dt) o[dt][r] *= alpha;
    }

    // re-layout P: C-format (lane=col) -> A-format (lane=row) via LDS bounce;
    // LDS ops of one wave are in-order, so no barrier needed, just a fence.
    unsigned short* pw = &Ps[w * 16 * 40];
#pragma unroll
    for (int r = 0; r < 8; ++r) {
      pw[(r + 8 * hi) * 40 + lo]      = f32_to_bf16(pr0[r]);
      pw[(r + 8 * hi) * 40 + 16 + lo] = f32_to_bf16(pr1[r]);
    }
    asm volatile("" ::: "memory");
    __builtin_amdgcn_wave_barrier();

    BF16Frag pa;
    const unsigned short* prow = &Ps[w * 16 * 40 + lo * 40];
    pa.h[0] = ld8u(prow + hi * 8);
    pa.h[1] = ld8u(prow + 16 + hi * 8);

    // O(16x128) += P(16x32) * V(32x128), 8 d-tiles of 16
#pragma unroll
    for (int dt = 0; dt < 8; ++dt) {
      const unsigned short* vp = &Vt[(dt * 16 + lo) * 40 + hi * 8];
      BF16Frag vbf;
      vbf.h[0] = ld8u(vp);
      vbf.h[1] = ld8u(vp + 16);
      o[dt] = wmma_bf16(pa.v, vbf.v, o[dt]);
    }
  }

  // normalize and store attention output (bf16) as [S, NH*HD]
#pragma unroll
  for (int r = 0; r < 8; ++r) {
    const float inv = 1.0f / lrun[r];
    const int   row = qrow0 + r + 8 * hi;
#pragma unroll
    for (int dt = 0; dt < 8; ++dt) {
      O[(size_t)row * (NH * HD) + h * HD + dt * 16 + lo] = f32_to_bf16(o[dt][r] * inv);
    }
  }
}

// ---------------------------------------------------------------------------
// Launch: convert once to bf16 (weight scratch reused across the 4 GEMMs),
// QKV projections -> flash attention -> output projection.
// Workspace: Xb 16MB + Wscratch 32MB + Q 16MB + K 4MB + V 4MB + attn 16MB = 88MB.
// The additive causal mask input (d_in[1]) is implemented analytically.
// ---------------------------------------------------------------------------
extern "C" void kernel_launch(void* const* d_in, const int* in_sizes, int n_in,
                              void* d_out, int out_size, void* d_ws, size_t ws_size,
                              hipStream_t stream) {
  (void)in_sizes; (void)n_in; (void)out_size; (void)ws_size;
  const float* X  = (const float*)d_in[0];
  const float* wq = (const float*)d_in[2];
  const float* wk = (const float*)d_in[3];
  const float* wv = (const float*)d_in[4];
  const float* wo = (const float*)d_in[5];

  unsigned short* Xb = (unsigned short*)d_ws;
  unsigned short* Wb = Xb + (size_t)S_LEN * HID;            // 32MB scratch (max weight)
  unsigned short* Qb = Wb + (size_t)HID * HID;
  unsigned short* Kb = Qb + (size_t)S_LEN * NH * HD;
  unsigned short* Vb = Kb + (size_t)S_LEN * NKV * HD;
  unsigned short* Ab = Vb + (size_t)S_LEN * NKV * HD;

  const int nX  = S_LEN * HID;          // 8.4M
  const int nWq = NH * HD * HID;        // 16.8M
  const int nWk = NKV * HD * HID;       // 4.2M

  cvt_f32_bf16<<<nX / 2048, 256, 0, stream>>>(X, Xb, nX);

  // Q projection
  cvt_f32_bf16<<<nWq / 2048, 256, 0, stream>>>(wq, Wb, nWq);
  gemm_bf16_nt<false><<<(S_LEN / 256) * (NH * HD / 64), 256, 0, stream>>>(Xb, Wb, Qb, S_LEN, NH * HD);
  // K projection
  cvt_f32_bf16<<<nWk / 2048, 256, 0, stream>>>(wk, Wb, nWk);
  gemm_bf16_nt<false><<<(S_LEN / 256) * (NKV * HD / 64), 256, 0, stream>>>(Xb, Wb, Kb, S_LEN, NKV * HD);
  // V projection
  cvt_f32_bf16<<<nWk / 2048, 256, 0, stream>>>(wv, Wb, nWk);
  gemm_bf16_nt<false><<<(S_LEN / 256) * (NKV * HD / 64), 256, 0, stream>>>(Xb, Wb, Vb, S_LEN, NKV * HD);

  attn_fwd<<<dim3(S_LEN / 128, NH), 256, 0, stream>>>(Qb, Kb, Vb, Ab);

  // Output projection (f32 result)
  cvt_f32_bf16<<<nWq / 2048, 256, 0, stream>>>(wo, Wb, nWq);
  gemm_bf16_nt<true><<<(S_LEN / 256) * (NH * HD / 64), 256, 0, stream>>>(Ab, Wb, d_out, S_LEN, NH * HD);
}
